// SpatialAttentionBlock_2173253452047
// MI455X (gfx1250) — matrix operand order repaired and checked by reading, not compiled, not run on hardware
//
#include <hip/hip_runtime.h>

// ---------------------------------------------------------------------------
// SpatialAttentionBlock for MI455X (gfx1250, wave32, WMMA).
// B=4, C=256, C8=32, H=W=64, HW=4096.
//
// Stage 1: qk_prep  -> Q,K bf16 [b][pos][32]   (conv + folded BN + ReLU)
// Stage 2: v_prep   -> Vt bf16 [b][c][pos]     (1x1 conv, 32 c per block)
// Stage 3: attn     -> flash-style fused softmax(Q^T K) @ V^T, WMMA bf16,
//                      64-wide m chunks, wave-uniform rescale skip
// ---------------------------------------------------------------------------

typedef __attribute__((ext_vector_type(16))) __bf16 v16bf;
typedef __attribute__((ext_vector_type(8)))  float  v8f;

union Frag16 { v16bf v; uint4 q[2]; };

#define SWZ_F(val, mask) \
  __int_as_float(__builtin_amdgcn_ds_swizzle(__float_as_int(val), ((mask) << 10) | 0x1f))

__device__ __forceinline__ float redmax16(float v) {
  v = fmaxf(v, SWZ_F(v, 1));
  v = fmaxf(v, SWZ_F(v, 2));
  v = fmaxf(v, SWZ_F(v, 4));
  v = fmaxf(v, SWZ_F(v, 8));
  return v;
}
__device__ __forceinline__ float redsum16(float v) {
  v += SWZ_F(v, 1);
  v += SWZ_F(v, 2);
  v += SWZ_F(v, 4);
  v += SWZ_F(v, 8);
  return v;
}

// ---------------------------------------------------------------------------
// Stage 1: Q = relu(BN(conv1x3(x))), K = relu(BN(conv3x1(x))); bf16 [b][pos][32]
// ---------------------------------------------------------------------------
__global__ __launch_bounds__(256) void qk_prep(
    const float* __restrict__ x,
    const float* __restrict__ Wq, const float* __restrict__ bq,
    const float* __restrict__ qs, const float* __restrict__ qb,
    const float* __restrict__ qm, const float* __restrict__ qv,
    const float* __restrict__ Wk, const float* __restrict__ bk,
    const float* __restrict__ ks, const float* __restrict__ kb,
    const float* __restrict__ km, const float* __restrict__ kv,
    __bf16* __restrict__ Q, __bf16* __restrict__ K)
{
  __shared__ float sWq[32][96];  // [o][ci_local*3 + tap], 12KB
  __shared__ float sWk[32][96];

  const int tid = threadIdx.x;
  const int gid = blockIdx.x * 256 + tid;   // b*4096 + hw
  const int b  = gid >> 12;
  const int hw = gid & 4095;
  const int h  = hw >> 6;
  const int w  = hw & 63;

  float accq[32], acck[32];
#pragma unroll
  for (int o = 0; o < 32; ++o) { accq[o] = 0.f; acck[o] = 0.f; }

  for (int cc = 0; cc < 8; ++cc) {          // 8 chunks of 32 input channels
    __syncthreads();
    for (int i = tid; i < 32 * 32 * 3; i += 256) {
      const int o = i / 96, rem = i % 96, cil = rem / 3, t = rem % 3;
      sWq[o][cil * 3 + t] = Wq[(o * 256 + cc * 32 + cil) * 3 + t];
      sWk[o][cil * 3 + t] = Wk[(o * 256 + cc * 32 + cil) * 3 + t];
    }
    __syncthreads();

    for (int cil = 0; cil < 32; ++cil) {
      const int ci = cc * 32 + cil;
      const float* xb = x + ((size_t)(b * 256 + ci) << 12) + hw;
      const float x0 = xb[0];
      const float xm = (w > 0)  ? xb[-1]  : 0.f;
      const float xp = (w < 63) ? xb[1]   : 0.f;
      const float ym = (h > 0)  ? xb[-64] : 0.f;
      const float yp = (h < 63) ? xb[64]  : 0.f;
#pragma unroll
      for (int o = 0; o < 32; ++o) {
        accq[o] += xm * sWq[o][cil * 3] + x0 * sWq[o][cil * 3 + 1] + xp * sWq[o][cil * 3 + 2];
        acck[o] += ym * sWk[o][cil * 3] + x0 * sWk[o][cil * 3 + 1] + yp * sWk[o][cil * 3 + 2];
      }
    }
  }

#pragma unroll
  for (int o = 0; o < 32; ++o) {
    const float invq = qs[o] * rsqrtf(qv[o] + 1e-5f);
    float vq = (accq[o] + bq[o]) * invq + (qb[o] - qm[o] * invq);
    Q[(size_t)gid * 32 + o] = (__bf16)fmaxf(vq, 0.f);
    const float invk = ks[o] * rsqrtf(kv[o] + 1e-5f);
    float vk = (acck[o] + bk[o]) * invk + (kb[o] - km[o] * invk);
    K[(size_t)gid * 32 + o] = (__bf16)fmaxf(vk, 0.f);
  }
}

// ---------------------------------------------------------------------------
// Stage 2: V = conv1x1(x) -> bf16 [b][c][pos].
// Block = (b, 32-wide c tile, 256-wide hw tile): x read once per 32 outputs
// (~128 MB total L2 traffic instead of 4 GB with 1 c per block).
// ---------------------------------------------------------------------------
__global__ __launch_bounds__(256) void v_prep(
    const float* __restrict__ x, const float* __restrict__ Wv,
    const float* __restrict__ bv, __bf16* __restrict__ Vt)
{
  const int hwt = blockIdx.x & 15;          // 16 hw tiles of 256
  const int ct  = (blockIdx.x >> 4) & 7;    // 8 c tiles of 32
  const int b   = blockIdx.x >> 7;
  const int tid = threadIdx.x;

  __shared__ float sw[32][256];             // 32KB of weights
  for (int i = tid; i < 32 * 256; i += 256) {
    const int co = i >> 8, ci = i & 255;
    sw[co][ci] = Wv[(ct * 32 + co) * 256 + ci];
  }
  __syncthreads();

  const int hw = hwt * 256 + tid;
  const float* xb = x + ((size_t)b << 20) + hw;
  float acc[32];
#pragma unroll
  for (int co = 0; co < 32; ++co) acc[co] = bv[ct * 32 + co];

  for (int ci = 0; ci < 256; ++ci) {
    const float xv = xb[(size_t)ci << 12];  // coalesced across lanes
#pragma unroll
    for (int co = 0; co < 32; ++co) acc[co] += sw[co][ci] * xv;  // LDS broadcast
  }

#pragma unroll
  for (int co = 0; co < 32; ++co)
    Vt[(((size_t)b * 256 + ct * 32 + co) << 12) + hw] = (__bf16)acc[co];
}

// ---------------------------------------------------------------------------
// Stage 3: fused attention. One wave per 16-query block, m streamed in chunks
// of 64: 4 S-WMMAs + 32 P.V-WMMAs per iteration. Online softmax with
// ds_swizzle row reductions; accumulator rescale skipped via wave-uniform
// branch once the running max stops changing (scalar branch, EXEC stays ~0).
// ---------------------------------------------------------------------------
__global__ __launch_bounds__(32) void attn_kernel(
    const __bf16* __restrict__ Q, const __bf16* __restrict__ K,
    const __bf16* __restrict__ Vt, const float* __restrict__ x,
    const float* __restrict__ gamma_p, float* __restrict__ out)
{
  const int lane = threadIdx.x & 31;
  const int half = lane >> 4;
  const int ln   = lane & 15;
  const int b    = blockIdx.x >> 8;
  const int n0   = (blockIdx.x & 255) << 4;

  __shared__ __align__(16) __bf16 Pt[16 * 64];  // P staging, 2KB

  // A-fragment of Q (row = ln, K slots 16*(v>>2)+8*half+2*(v&3)+p)
  Frag16 aQ;
  {
    const char* qrow = (const char*)(Q + (size_t)(b * 4096 + n0 + ln) * 32);
    aQ.q[0] = *(const uint4*)(qrow + 16 * half);
    aQ.q[1] = *(const uint4*)(qrow + 32 + 16 * half);
  }

  v8f vzero = {};
  v8f acc[16];                 // 16x256 fp32 output tile (128 VGPRs)
#pragma unroll
  for (int t = 0; t < 16; ++t) acc[t] = vzero;

  float mrow[8], lrow[8];      // per-lane state for rows r + 8*half
#pragma unroll
  for (int r = 0; r < 8; ++r) { mrow[r] = -__builtin_inff(); lrow[r] = 0.f; }

  const char* kbase = (const char*)(K + (size_t)b * 4096 * 32);
  const char* vbase = (const char*)(Vt + (size_t)b * 256 * 4096);

  for (int mc = 0; mc < 64; ++mc) {
    const int m0 = mc << 6;

    // 4 S tiles: S[j] covers m columns [m0+16j, m0+16j+16)
    v8f Sx[4];
#pragma unroll
    for (int j = 0; j < 4; ++j) {
      Frag16 bK;
      const char* kr = kbase + (size_t)(m0 + j * 16 + ln) * 64 + 16 * half;
      bK.q[0] = *(const uint4*)(kr);
      bK.q[1] = *(const uint4*)(kr + 32);
      Sx[j] = __builtin_amdgcn_wmma_f32_16x16x32_bf16(false, aQ.v, false, bK.v,
                                                      (short)0, vzero, false, false);
    }

    // online softmax per row (C/D layout: row = r + 8*half, col = 16j + ln)
    float scr[8];
    int chg = 0;
#pragma unroll
    for (int r = 0; r < 8; ++r) {
      float cm = fmaxf(fmaxf(Sx[0][r], Sx[1][r]), fmaxf(Sx[2][r], Sx[3][r]));
      cm = redmax16(cm);
      const float mn = fmaxf(mrow[r], cm);
      chg |= (mn > mrow[r]) ? 1 : 0;
      scr[r] = __expf(mrow[r] - mn);
      mrow[r] = mn;
      const float p0 = __expf(Sx[0][r] - mn);
      const float p1 = __expf(Sx[1][r] - mn);
      const float p2 = __expf(Sx[2][r] - mn);
      const float p3 = __expf(Sx[3][r] - mn);
      lrow[r] = lrow[r] * scr[r] + (p0 + p1 + p2 + p3);
      const int row = r + 8 * half;
      Pt[row * 64 + ln]      = (__bf16)p0;
      Pt[row * 64 + 16 + ln] = (__bf16)p1;
      Pt[row * 64 + 32 + ln] = (__bf16)p2;
      Pt[row * 64 + 48 + ln] = (__bf16)p3;
    }

    // rescale accumulators only if any row max advanced (wave-uniform branch)
    if (__any(chg)) {
#pragma unroll
      for (int t = 0; t < 16; ++t)
#pragma unroll
        for (int r = 0; r < 8; ++r) acc[t][r] *= scr[r];
    }
    __syncthreads();

    // Re-read P as two A fragments (16x32 halves of the 16x64 tile)
    Frag16 aP0, aP1;
    {
      const char* prow = (const char*)(Pt + ln * 64);
      aP0.q[0] = *(const uint4*)(prow + 16 * half);
      aP0.q[1] = *(const uint4*)(prow + 32 + 16 * half);
      aP1.q[0] = *(const uint4*)(prow + 64 + 16 * half);
      aP1.q[1] = *(const uint4*)(prow + 96 + 16 * half);
    }
    __syncthreads();

    // O += P(16x64) @ V^T(64x256): 16 column tiles x 2 K-steps
#pragma unroll
    for (int t = 0; t < 16; ++t) {
      Frag16 bV0, bV1;
      const char* vrow = vbase + ((size_t)(t * 16 + ln) * 4096 + m0 + 8 * half) * 2;
      bV0.q[0] = *(const uint4*)(vrow);
      bV0.q[1] = *(const uint4*)(vrow + 32);
      bV1.q[0] = *(const uint4*)(vrow + 64);
      bV1.q[1] = *(const uint4*)(vrow + 96);
      acc[t] = __builtin_amdgcn_wmma_f32_16x16x32_bf16(false, aP0.v, false, bV0.v,
                                                       (short)0, acc[t], false, false);
      acc[t] = __builtin_amdgcn_wmma_f32_16x16x32_bf16(false, aP1.v, false, bV1.v,
                                                       (short)0, acc[t], false, false);
    }
  }

  // epilogue: normalize, scale by gamma, add residual, scatter to NCHW
  const float g = gamma_p[0];
#pragma unroll
  for (int r = 0; r < 8; ++r) {
    const float inv = 1.f / redsum16(lrow[r]);
    const int n = n0 + r + 8 * half;
#pragma unroll
    for (int t = 0; t < 16; ++t) {
      const int c = t * 16 + ln;
      const size_t idx = (((size_t)b * 256 + c) << 12) + n;
      out[idx] = g * (acc[t][r] * inv) + x[idx];
    }
  }
}

// ---------------------------------------------------------------------------
extern "C" void kernel_launch(void* const* d_in, const int* in_sizes, int n_in,
                              void* d_out, int out_size, void* d_ws, size_t ws_size,
                              hipStream_t stream) {
  const float* x  = (const float*)d_in[0];
  const float* Wq = (const float*)d_in[1];
  const float* bq = (const float*)d_in[2];
  const float* qs = (const float*)d_in[3];
  const float* qb = (const float*)d_in[4];
  const float* qm = (const float*)d_in[5];
  const float* qv = (const float*)d_in[6];
  const float* Wk = (const float*)d_in[7];
  const float* bk = (const float*)d_in[8];
  const float* ks = (const float*)d_in[9];
  const float* kb = (const float*)d_in[10];
  const float* km = (const float*)d_in[11];
  const float* kv = (const float*)d_in[12];
  const float* Wv = (const float*)d_in[13];
  const float* bv = (const float*)d_in[14];
  const float* gm = (const float*)d_in[15];
  float* out = (float*)d_out;

  // workspace: Q 1MB | K 1MB | Vt 8MB (all bf16)
  const size_t QK_ELEMS = (size_t)4 * 4096 * 32;
  __bf16* Q  = (__bf16*)d_ws;
  __bf16* K  = (__bf16*)((char*)d_ws + QK_ELEMS * 2);
  __bf16* Vt = (__bf16*)((char*)d_ws + 2 * QK_ELEMS * 2);

  qk_prep<<<dim3(64), dim3(256), 0, stream>>>(x, Wq, bq, qs, qb, qm, qv,
                                              Wk, bk, ks, kb, km, kv, Q, K);
  v_prep<<<dim3(512), dim3(256), 0, stream>>>(x, Wv, bv, Vt);
  attn_kernel<<<dim3(1024), dim3(32), 0, stream>>>(Q, K, Vt, x, gm, out);
}
